// BondCenteredTensorMomentDescriptor_57913339019924
// MI455X (gfx1250) — matrix operand order repaired and checked by reading, not compiled, not run on hardware
//
#include <hip/hip_runtime.h>
#include <math.h>

// ---------------------------------------------------------------------------
// BondCenteredTensorMomentDescriptor for MI455X (gfx1250, wave32, WMMA).
//
//  out[e,k3,f] = rad[e,f]*cut[e] * ( M_e @ X_e )[k3,f]
//    M_e[k3,(p,m1)] = sum_m2 cg_p[m1,m2,k3] * Ysph_e[m2]      (32 x 256, f16)
//    X_e[(p,m1),f]  = y_e[m1,f] * w2[p,f]                     (256 x 16, f16)
//    y_e[k,f]       = sum_nz cg * a_i[i,f]*a_j[j,f]*w1[p,f]   (sparse, f32 VALU)
//
// CDNA5 features used:
//   - v_wmma_f32_16x16x32_f16 (16 per edge) for the TP2 GEMM
//   - global_load_async_to_lds_b128 + s_wait_asynccnt for the edge gather
//   - tensor_load_to_lds (TDM, 6-arg clang-23 form) + s_wait_tensorcnt
//   - LDS phase overlay: 24.7KB/wave -> ~12 waves/WGP to hide ds latency
// ---------------------------------------------------------------------------

#define NPATH   42
#define NK      25
#define KCOLS   230
#define KPAD    256
#define TP1CAP  8192
#define TP2CAP  8192
#define CELLCAP 4096
#define CUTOFF_F 5.0f
#define PI_F 3.14159265358979323846f

typedef __attribute__((ext_vector_type(16))) _Float16 v16h;
typedef __attribute__((ext_vector_type(8)))  _Float16 h8;
typedef __attribute__((ext_vector_type(8)))  float    v8f;
typedef unsigned int u32x4 __attribute__((ext_vector_type(4)));
typedef int          i32x8 __attribute__((ext_vector_type(8)));
typedef int          i32x4 __attribute__((ext_vector_type(4)));

// ---------------- LDS arena (phase overlay) --------------------------------
// persistent: sX (B operand) + sSph
// phase A  : sA (gathered descriptors), sY (TP1 partials), sW1, sW2
// phase B  : sM (A operand) overlays phase A
#define OFF_X   0        // 16*256 f16          = 8192 B
#define OFF_SPH 8192     // 32 f32              = 128 B
#define OFF_A   8320     // 2*25*16 f32         = 3200 B
#define OFF_Y   11520    // 2*25*16 f32         = 3200 B
#define OFF_W1  14720    // 42*16 f32           = 2688 B
#define OFF_W2  17408    // 42*16 f32           = 2688 B   (phase A ends 20096)
#define OFF_M   8320     // 32*256 f16          = 16384 B  (phase B ends 24704)
#define SMEM_BYTES 24704

// ---------------- path enumeration (must match Python CG_PATHS order) ------
__host__ __device__ inline void get_path(int p, int* L1, int* L2, int* L3) {
  int c = 0;
  for (int l1 = 0; l1 <= 4; ++l1)
    for (int l2 = 0; l2 <= 4; ++l2) {
      int lo = l1 > l2 ? l1 - l2 : l2 - l1;
      int hi = (l1 + l2 < 4) ? (l1 + l2) : 4;
      for (int l3 = lo; l3 <= hi; ++l3)
        if (((l1 + l2 + l3) & 1) == 0) {
          if (c == p) { *L1 = l1; *L2 = l2; *L3 = l3; return; }
          ++c;
        }
    }
  *L1 = 0; *L2 = 0; *L3 = 0;
}

// ---------------- setup kernel 1: dense real CG tensors --------------------
struct cdx { double re, im; };
__device__ inline cdx cmul(cdx a, cdx b) { return { a.re*b.re - a.im*b.im, a.re*b.im + a.im*b.re }; }

__device__ inline double dfact(int n) { double r = 1.0; for (int t = 2; t <= n; ++t) r *= (double)t; return r; }

__device__ double cg_complex(int j1, int m1, int j2, int m2, int j3, int m3) {
  int dj = j1 > j2 ? j1 - j2 : j2 - j1;
  if (m1 + m2 != m3 || j3 < dj || j3 > j1 + j2) return 0.0;
  double pre = sqrt((double)(2*j3+1) * dfact(j1+j2-j3) * dfact(j1-j2+j3) * dfact(-j1+j2+j3) / dfact(j1+j2+j3+1));
  pre *= sqrt(dfact(j1+m1)*dfact(j1-m1)*dfact(j2+m2)*dfact(j2-m2)*dfact(j3+m3)*dfact(j3-m3));
  double s = 0.0;
  for (int k = 0; k <= j1 + j2 + j3; ++k) {
    int t0 = k, t1 = j1+j2-j3-k, t2 = j1-m1-k, t3 = j2+m2-k, t4 = j3-j2+m1+k, t5 = j3-j1-m2+k;
    if (t0 < 0 || t1 < 0 || t2 < 0 || t3 < 0 || t4 < 0 || t5 < 0) continue;
    double term = 1.0 / (dfact(t0)*dfact(t1)*dfact(t2)*dfact(t3)*dfact(t4)*dfact(t5));
    s += (k & 1) ? -term : term;
  }
  return pre * s;
}

// Row 'a' of the complex->real change-of-basis U(l): at most 2 nonzeros.
__device__ inline int rowU(int l, int a, int* col, cdx* val) {
  const double is2 = 0.70710678118654752440;
  int m = a - l;
  if (m == 0) { col[0] = l; val[0] = {1.0, 0.0}; return 1; }
  if (m > 0) {
    col[0] = l + m; val[0] = { (m & 1) ? -is2 : is2, 0.0 };
    col[1] = l - m; val[1] = { is2, 0.0 };
    return 2;
  }
  int mm = -m;
  col[0] = l - mm; val[0] = { 0.0, is2 };
  col[1] = l + mm; val[1] = { 0.0, (mm & 1) ? is2 : -is2 };  // -i*(-1)^m/sqrt2
  return 2;
}

__global__ void setup_cg_kernel(float* __restrict__ cg) {
  int p = blockIdx.x;
  if (p >= NPATH) return;
  int l1, l2, l3; get_path(p, &l1, &l2, &l3);
  int d2 = 2*l2+1, d3 = 2*l3+1, n = (2*l1+1)*d2*d3;
  for (int idx = threadIdx.x; idx < n; idx += blockDim.x) {
    int a  = idx / (d2*d3);
    int rm = idx - a*(d2*d3);
    int b  = rm / d3;
    int cc = rm - b*d3;
    int ca[2], cb[2], ck[2]; cdx va[2], vb[2], vk[2];
    int na = rowU(l1, a,  ca, va);
    int nb = rowU(l2, b,  cb, vb);
    int nc = rowU(l3, cc, ck, vk);
    double acc = 0.0;
    for (int ia = 0; ia < na; ++ia)
      for (int ib = 0; ib < nb; ++ib)
        for (int ic = 0; ic < nc; ++ic) {
          double cgv = cg_complex(l1, ca[ia]-l1, l2, cb[ib]-l2, l3, ck[ic]-l3);
          if (cgv == 0.0) continue;
          cdx u = cmul(va[ia], vb[ib]);
          cdx w = { vk[ic].re, -vk[ic].im };          // conj(U3)
          cdx t = cmul(u, w);
          acc += t.re * cgv;                          // result is real
        }
    cg[p*729 + idx] = (float)acc;
  }
}

// ---------------- setup kernel 2: sparse lists ------------------------------
__global__ void setup_lists_kernel(const float* __restrict__ cg,
                                   int* __restrict__ tp1_kstart, uint2* __restrict__ tp1,
                                   int* __restrict__ meta, uint2* __restrict__ cells,
                                   uint2* __restrict__ tp2e,
                                   int* __restrict__ kcol_src, int* __restrict__ kcol_p) {
  if (threadIdx.x != 0 || blockIdx.x != 0) return;
  // ---- TP1 entries, grouped by global output index k3g ----
  int n1 = 0;
  for (int kg = 0; kg < NK; ++kg) {
    tp1_kstart[kg] = n1;
    int l3k = (kg >= 16) ? 4 : (kg >= 9) ? 3 : (kg >= 4) ? 2 : (kg >= 1) ? 1 : 0;
    int kl = kg - l3k*l3k;
    for (int p = 0; p < NPATH; ++p) {
      int l1, l2, l3; get_path(p, &l1, &l2, &l3);
      if (l3 != l3k) continue;
      int d2 = 2*l2+1, d3 = 2*l3+1;
      for (int i = 0; i < 2*l1+1; ++i)
        for (int j = 0; j < d2; ++j) {
          float v = cg[p*729 + (i*d2 + j)*d3 + kl];
          if (fabsf(v) > 1e-8f && n1 < TP1CAP) {
            uint2 en;
            en.x = (unsigned)(i + l1*l1) | ((unsigned)(j + l2*l2) << 8) | ((unsigned)p << 16);
            en.y = __float_as_uint(v);
            tp1[n1++] = en;
          }
        }
    }
  }
  tp1_kstart[NK] = n1;
  // ---- X column tables: (p, m1) -> source k1 index and path ----
  int ncol = 0;
  for (int p = 0; p < NPATH; ++p) {
    int l1, l2, l3; get_path(p, &l1, &l2, &l3);
    for (int m1 = 0; m1 < 2*l1+1; ++m1) { kcol_src[ncol] = l1*l1 + m1; kcol_p[ncol] = p; ++ncol; }
  }
  for (int c = ncol; c < KPAD; ++c) { kcol_src[c] = 0; kcol_p[c] = 0; }
  // ---- TP2: per (kcol, k3) cell -> list of (m2_global, cg) ----
  int ncell = 0, ne = 0, c = 0;
  for (int p = 0; p < NPATH; ++p) {
    int l1, l2, l3; get_path(p, &l1, &l2, &l3);
    int d2 = 2*l2+1, d3 = 2*l3+1;
    for (int m1 = 0; m1 < 2*l1+1; ++m1, ++c) {
      for (int k = 0; k < d3; ++k) {
        int start = ne, cnt = 0;
        for (int m2 = 0; m2 < d2; ++m2) {
          float v = cg[p*729 + (m1*d2 + m2)*d3 + k];
          if (fabsf(v) > 1e-8f && ne < TP2CAP) {
            uint2 e2; e2.x = (unsigned)(l2*l2 + m2); e2.y = __float_as_uint(v);
            tp2e[ne++] = e2; ++cnt;
          }
        }
        if (cnt && ncell < CELLCAP) {
          uint2 cl;
          cl.x = (unsigned)c | ((unsigned)(l3*l3 + k) << 8);
          cl.y = (unsigned)start | ((unsigned)cnt << 20);
          cells[ncell++] = cl;
        }
      }
    }
  }
  meta[0] = ncell;
}

// ---------------- CDNA5 data movement helpers -------------------------------
// Async global->LDS copy of one 16B chunk (ASYNCcnt-tracked, no VGPR staging).
__device__ inline void async_gather16(unsigned lds_dst, const void* gsrc) {
  asm volatile("global_load_async_to_lds_b128 %0, %1, off"
               :: "v"(lds_dst), "v"((unsigned long long)(uintptr_t)gsrc)
               : "memory");
}
__device__ inline void wait_asynccnt0() {
  asm volatile("s_wait_asynccnt 0x0" ::: "memory");
}

// TDM 1-D tensor load: n 4-byte elements from global -> LDS (TENSORcnt).
// D# per ISA ch.8: group0 = {count|flags, lds_addr, global_addr, type=2},
// group1 = {data_size=4B, dims/strides}, groups 2/3/4 zero (<=2D tensor).
__device__ inline void tdm_load_1d(unsigned lds_off, const void* gptr, int n) {
  unsigned long long ga = (unsigned long long)(uintptr_t)gptr;
  u32x4 g0;
  g0.x = 1u;                                              // count=1, user mode
  g0.y = lds_off;                                         // lds_addr [63:32]
  g0.z = (unsigned)(ga & 0xFFFFFFFFu);                    // global_addr lo
  g0.w = (unsigned)((ga >> 32) & 0x01FFFFFFu) | (2u << 30); // addr hi | type=2
  i32x8 g1;
  g1[0] = (2 << 16);                                      // data_size = 4B
  g1[1] = (int)(((unsigned)n & 0xFFFFu) << 16);           // tensor_dim0 lo16
  g1[2] = (int)((((unsigned)n >> 16) & 0xFFFFu) | (1u << 16)); // dim0 hi | tensor_dim1=1
  g1[3] = (int)(((unsigned)n & 0xFFFFu) << 16);           // tile_dim0 = n
  g1[4] = 0;                                              // tile_dim1/2 unused
  g1[5] = n;                                              // tensor_dim0_stride lo32
  g1[6] = 0;
  g1[7] = 0;
  i32x4 g2 = {0, 0, 0, 0};
  i32x4 g3 = {0, 0, 0, 0};
  i32x8 g4 = {0, 0, 0, 0, 0, 0, 0, 0};
  __builtin_amdgcn_tensor_load_to_lds(g0, g1, g2, g3, g4, 0);
}

// ---------------- helpers for WMMA fragment loads ---------------------------
// A-matrix f16 16x32 layout (ISA 7.12.2): lanes 0-15 hold K{0-7,16-23},
// lanes 16-31 hold K{8-15,24-31} of row (lane&15).
__device__ inline v16h load_a_frag(const _Float16* row, int k0, int ksel) {
  h8 lo = *(const h8*)(row + k0 + ksel * 8);
  h8 hi = *(const h8*)(row + k0 + 16 + ksel * 8);
  v16h a;
#pragma unroll
  for (int t = 0; t < 8; ++t) { a[t] = lo[t]; a[t + 8] = hi[t]; }
  return a;
}

// ---------------- main per-edge kernel --------------------------------------
__global__ __launch_bounds__(32)
void edge_kernel(const float* __restrict__ desc, const float* __restrict__ disp,
                 const float* __restrict__ w1,   const float* __restrict__ w2,
                 const int* __restrict__ nbi,    const int* __restrict__ nbj,
                 float* __restrict__ out, int E,
                 const int* __restrict__ tp1_kstart, const uint2* __restrict__ tp1,
                 const int* __restrict__ meta,       const uint2* __restrict__ cells,
                 const uint2* __restrict__ tp2e,
                 const int* __restrict__ kcol_src,   const int* __restrict__ kcol_p) {
  __shared__ __align__(32) char smem[SMEM_BYTES];
  _Float16* sX   = (_Float16*)(smem + OFF_X);    // persistent: B operand (N x K)
  float*    sSph = (float*)   (smem + OFF_SPH);  // persistent: sph harmonics
  float*    sA   = (float*)   (smem + OFF_A);    // phase A: gathered a_i/a_j
  float*    sY   = (float*)   (smem + OFF_Y);    // phase A: TP1 partials
  float*    sW1  = (float*)   (smem + OFF_W1);   // phase A: weights
  float*    sW2  = (float*)   (smem + OFF_W2);
  _Float16* sM   = (_Float16*)(smem + OFF_M);    // phase B: A operand (M x K)

  const int lane = threadIdx.x;
  const int f    = lane & 15;   // feature / N column
  const int half = lane >> 4;   // lane group
  int e = blockIdx.x; if (e >= E) e = E - 1;

  // ---- weights via Tensor Data Mover (one TDM op each, TENSORcnt) --------
  tdm_load_1d((unsigned)(uintptr_t)(smem + OFF_W1), w1, NPATH * 16);
  tdm_load_1d((unsigned)(uintptr_t)(smem + OFF_W2), w2, NPATH * 16);

  // ---- descriptor gather via async global->LDS (ASYNCcnt) ----------------
  const int ni = nbi[e], nj = nbj[e];
  const char* A1b = (const char*)(desc + (size_t)ni * (NK * 16));
  const char* A2b = (const char*)(desc + (size_t)nj * (NK * 16));
  const unsigned sA_lds = (unsigned)(uintptr_t)(smem + OFF_A);
  for (int t = lane; t < 200; t += 32) {                 // 200 x 16B = 3200B
    const char* src = (t < 100) ? (A1b + t * 16) : (A2b + (t - 100) * 16);
    async_gather16(sA_lds + t * 16, src);
  }
  wait_asynccnt0();
  __builtin_amdgcn_s_wait_tensorcnt(0);
  __syncthreads();

  // ---- TP1: sparse CG contraction, entries split across the two halves ----
  for (int k = 0; k < NK; ++k) {
    int s0 = tp1_kstart[k], s1 = tp1_kstart[k + 1];
    float acc = 0.f;
    for (int t = s0 + half; t < s1; t += 2) {
      uint2 en = tp1[t];
      int i = en.x & 255, j = (en.x >> 8) & 255, p = en.x >> 16;
      acc = fmaf(__uint_as_float(en.y) * sW1[p * 16 + f],
                 sA[i * 16 + f] * sA[(25 + j) * 16 + f], acc);
    }
    sY[(half * NK + k) * 16 + f] = acc;
  }

  // ---- geometry: radial basis + cutoff (per-lane f), sph harmonics --------
  float dx = disp[3*e], dy = disp[3*e+1], dz = disp[3*e+2];
  float r = sqrtf(dx*dx + dy*dy + dz*dz + 1e-12f);
  float inv = 1.f / r;
  float ux = dx*inv, uy = dy*inv, uz = dz*inv;
  float xx = r * (float)(f + 1) * (1.f / CUTOFF_F);
  float radf = (xx == 0.f) ? 1.f : sinf(PI_F * xx) / (PI_F * xx);
  float cutf = (r < CUTOFF_F) ? 0.5f * (1.f + cosf(PI_F * r / CUTOFF_F)) : 0.f;
  float scale = radf * cutf;

  if (lane == 0) {
    float Cm[5], Sm[5]; Cm[0] = 1.f; Sm[0] = 0.f;
#pragma unroll
    for (int m = 1; m <= 4; ++m) {
      Cm[m] = ux * Cm[m-1] - uy * Sm[m-1];
      Sm[m] = ux * Sm[m-1] + uy * Cm[m-1];
    }
    float Q[5][5];
    Q[0][0] = 1.f;
#pragma unroll
    for (int m = 0; m <= 4; ++m) {
      if (m > 0) Q[m][m] = (float)(2*m - 1) * Q[m-1][m-1];
      if (m + 1 <= 4) Q[m+1][m] = (float)(2*m + 1) * uz * Q[m][m];
#pragma unroll
      for (int l = m + 2; l <= 4; ++l)
        Q[l][m] = ((float)(2*l - 1) * uz * Q[l-1][m] - (float)(l + m - 1) * Q[l-2][m]) / (float)(l - m);
    }
    const float FOURPI = 12.5663706143591729539f;
    const float fct[9] = {1.f, 1.f, 2.f, 6.f, 24.f, 120.f, 720.f, 5040.f, 40320.f};
#pragma unroll
    for (int l = 0; l <= 4; ++l) {
      sSph[l*l + l] = sqrtf((float)(2*l + 1) / FOURPI) * Q[l][0];
#pragma unroll
      for (int m = 1; m <= l; ++m) {
        float K = sqrtf((float)(2*l + 1) / FOURPI * fct[l - m] / fct[l + m]);
        float base = 1.41421356237f * K * Q[l][m];
        sSph[l*l + l + m] = base * Cm[m];
        sSph[l*l + l - m] = base * Sm[m];
      }
    }
  }
  __syncthreads();

  // ---- X = y (w2-scaled, f16), stored N-major for B fragments -------------
  for (int c = half; c < KCOLS; c += 2) {
    int src = kcol_src[c];
    sX[f * KPAD + c] =
        (_Float16)((sY[src * 16 + f] + sY[(NK + src) * 16 + f]) * sW2[kcol_p[c] * 16 + f]);
  }
  for (int c = KCOLS + half; c < KPAD; c += 2) sX[f * KPAD + c] = (_Float16)0.f;
  __syncthreads();   // phase A buffers dead; sM overlay may now be written

  // ---- M = cg x Ysph (f16), zero then sparse fill (one cell per lane) -----
  {
    float4 z4 = {0.f, 0.f, 0.f, 0.f};
    float4* row4 = (float4*)(sM + lane * KPAD);
#pragma unroll
    for (int t = 0; t < KPAD / 8; ++t) row4[t] = z4;   // 8 halves per float4
  }
  __syncthreads();
  int ncell = meta[0];
  for (int c = lane; c < ncell; c += 32) {
    uint2 cl = cells[c];
    int kcol = cl.x & 255, kk = cl.x >> 8;
    int st = cl.y & 0xFFFFF, cnt = (int)(cl.y >> 20);
    float acc = 0.f;
    for (int t = 0; t < cnt; ++t) {
      uint2 e2 = tp2e[st + t];
      acc = fmaf(__uint_as_float(e2.y), sSph[e2.x], acc);
    }
    sM[kk * KPAD + kcol] = (_Float16)acc;
  }
  __syncthreads();

  // ---- TP2 GEMM on matrix cores: D(32x16) = M(32x256) @ X(256x16) ---------
  v8f c0 = {}; v8f c1 = {};
  const int ksel = half;
#pragma unroll
  for (int kb = 0; kb < 8; ++kb) {
    int k0 = kb * 32;
    v16h a0 = load_a_frag(sM + f * KPAD,        k0, ksel);
    v16h a1 = load_a_frag(sM + (16 + f) * KPAD, k0, ksel);
    // B f16 32x16: lane (N=f) holds K{ksel*8..+7, 16+ksel*8..+7} (same order as A)
    v16h b = load_a_frag(sX + f * KPAD, k0, ksel);
    c0 = __builtin_amdgcn_wmma_f32_16x16x32_f16(false, a0, false, b, (short)0, c0, false, false);
    c1 = __builtin_amdgcn_wmma_f32_16x16x32_f16(false, a1, false, b, (short)0, c1, false, false);
  }

  // ---- store: C/D f32 layout -> out[e, k3, f], scaled by rad*cut ----------
  float* O = out + (size_t)e * (NK * 16);
#pragma unroll
  for (int rr = 0; rr < 8; ++rr) {
    int row0 = ksel * 8 + rr;
    O[row0 * 16 + f] = c0[rr] * scale;
    int row1 = 16 + ksel * 8 + rr;
    if (row1 < NK) O[row1 * 16 + f] = c1[rr] * scale;
  }
}

// ---------------- launcher ---------------------------------------------------
extern "C" void kernel_launch(void* const* d_in, const int* in_sizes, int n_in,
                              void* d_out, int out_size, void* d_ws, size_t ws_size,
                              hipStream_t stream) {
  const float* desc = (const float*)d_in[0];   // (N,1,25,16)
  const float* disp = (const float*)d_in[1];   // (E,3)
  const float* w1   = (const float*)d_in[2];   // (42,16)
  const float* w2   = (const float*)d_in[3];   // (42,16)
  const int*   nbi  = (const int*)d_in[4];     // (E,)
  const int*   nbj  = (const int*)d_in[5];     // (E,)
  int E = in_sizes[1] / 3;
  float* out = (float*)d_out;

  char* ws = (char*)d_ws;
  size_t o = 0;
  float* cg = (float*)(ws + o);        o += (size_t)NPATH * 729 * sizeof(float); o = (o + 15) & ~(size_t)15;
  int*   tp1_kstart = (int*)(ws + o);  o += 32 * sizeof(int);
  uint2* tp1 = (uint2*)(ws + o);       o += (size_t)TP1CAP * sizeof(uint2);
  int*   meta = (int*)(ws + o);        o += 16;
  uint2* cells = (uint2*)(ws + o);     o += (size_t)CELLCAP * sizeof(uint2);
  uint2* tp2e = (uint2*)(ws + o);      o += (size_t)TP2CAP * sizeof(uint2);
  int*   ksrc = (int*)(ws + o);        o += KPAD * sizeof(int);
  int*   kp   = (int*)(ws + o);        o += KPAD * sizeof(int);
  (void)ws_size; (void)n_in; (void)out_size;

  setup_cg_kernel<<<NPATH, 128, 0, stream>>>(cg);
  setup_lists_kernel<<<1, 1, 0, stream>>>(cg, tp1_kstart, tp1, meta, cells, tp2e, ksrc, kp);
  edge_kernel<<<E, 32, 0, stream>>>(desc, disp, w1, w2, nbi, nbj, out, E,
                                    tp1_kstart, tp1, meta, cells, tp2e, ksrc, kp);
}
